// CrissCrossAttention_369367187525
// MI455X (gfx1250) — matrix operand order repaired and checked by reading, not compiled, not run on hardware
//
#include <hip/hip_runtime.h>
#include <hip/hip_bf16.h>
#include <math.h>

// ---------------------------------------------------------------------------
// CrissCrossAttention on gfx1250 (MI455X): bf16 WMMA everywhere.
// B=8, C=512, H=W=96, CQ=64.
// qkv packed buffer: per batch, rows 0..63 = q, 64..127 = k, 128..639 = v.
// ---------------------------------------------------------------------------

typedef __attribute__((ext_vector_type(16))) __bf16 bf16x16;
typedef __attribute__((ext_vector_type(8)))  float  v8f;

union FragU { bf16x16 v; uint4 q[2]; };
union AccU  { v8f v; float f[8]; };

__device__ inline v8f wmma_bf16(bf16x16 a, bf16x16 b, v8f c) {
  // D = A(16x32) * B(32x16) + C(16x16 f32)
  return __builtin_amdgcn_wmma_f32_16x16x32_bf16(
      /*neg_a=*/false, a, /*neg_b=*/false, b,
      /*c_mod=*/(short)0, c, /*reuse_a=*/false, /*reuse_b=*/false);
}

// A operand fragment: LDS rows m (16 rows at `base`), row length `stride`,
// K slice at kc. A 16x32 layout (matching B): lanes 0-15 => M=lane,
// K=0..15; lanes 16-31: K=16..31.
__device__ inline bf16x16 load_frag_A(const __bf16* base, int stride, int kc, int lane) {
  int m = lane & 15, h = lane >> 4;
  const __bf16* row = base + m * stride + kc;
  FragU f;
  f.q[0] = *(const uint4*)(row + h * 16);
  f.q[1] = *(const uint4*)(row + h * 16 + 8);
  return f.v;
}

// B operand fragment from transposed LDS tile Bt[n][k]. 32x16 B layout, contiguous k (not ISA order):
// N=lane%16; lanes 0-15 hold K=0..15, lanes 16-31 K=16..31, VGPRj=K(2j),K(2j+1).
__device__ inline bf16x16 load_frag_B(const __bf16* base, int stride, int kc, int lane) {
  int n = lane & 15, h = lane >> 4;
  const __bf16* row = base + n * stride + kc + h * 16;
  FragU f;
  f.q[0] = *(const uint4*)(row);
  f.q[1] = *(const uint4*)(row + 8);
  return f.v;
}

// ---------------------------------------------------------------------------
// Kernel 0: pack Wq|Wk|Wv -> bf16 Wcat(640x512), biases -> f32 bcat(640)
// ---------------------------------------------------------------------------
__global__ __launch_bounds__(256) void pack_weights_k(
    const float* __restrict__ Wq, const float* __restrict__ Wk,
    const float* __restrict__ Wv, const float* __restrict__ bq,
    const float* __restrict__ bk, const float* __restrict__ bv,
    __bf16* __restrict__ Wcat, float* __restrict__ bcat) {
  int i = blockIdx.x * 256 + threadIdx.x;
  if (i < 640) {
    bcat[i] = (i < 64) ? bq[i] : (i < 128 ? bk[i - 64] : bv[i - 128]);
  }
  if (i >= 640 * 512) return;
  int r = i >> 9, c = i & 511;
  float v;
  if (r < 64)       v = Wq[r * 512 + c];
  else if (r < 128) v = Wk[(r - 64) * 512 + c];
  else              v = Wv[(r - 128) * 512 + c];
  Wcat[i] = (__bf16)v;
}

// ---------------------------------------------------------------------------
// Kernel 1: projection GEMM: qkv[b,o,p] = Wcat[o,:] . x[b,:,p] + bcat[o]
// M=640 (5 blocks of 128), N=9216 (96 blocks of 96), K=512 (16 steps of 32).
// 256 threads = 8 waves; wave w owns m-tile w, all 6 n-tiles.
// ---------------------------------------------------------------------------
__global__ __launch_bounds__(256) void proj_gemm_k(
    const float* __restrict__ x, const __bf16* __restrict__ Wcat,
    const float* __restrict__ bcat, __bf16* __restrict__ qkv) {
  const int b  = blockIdx.z;
  const int m0 = blockIdx.y * 128;
  const int p0 = blockIdx.x * 96;
  __shared__ __bf16 sA[128 * 40];   // A[m][k], stride 40
  __shared__ __bf16 sB[96 * 40];    // Bt[n][k], stride 40

  const int t = threadIdx.x, lane = t & 31, wv = t >> 5;
  const v8f zero = {0.f, 0.f, 0.f, 0.f, 0.f, 0.f, 0.f, 0.f};
  v8f acc[6];
  for (int i = 0; i < 6; i++) acc[i] = zero;

  const float* xb = x + (size_t)b * 512 * 9216;

  for (int kc = 0; kc < 512; kc += 32) {
    __syncthreads();
    // A tile: Wcat[m0+m][kc + q4*8 ..+7], vectorized 8 bf16 per load
    for (int i = 0; i < 2; i++) {
      int idx = t + i * 256;            // 512 uint4
      int m = idx >> 2, q4 = idx & 3;
      *(uint4*)(sA + m * 40 + q4 * 8) =
          *(const uint4*)(Wcat + (size_t)(m0 + m) * 512 + kc + q4 * 8);
    }
    // B tile: x[b][kc+k][p0 + 4n..], float4 loads, bf16 scatter into Bt[n][k]
    for (int i = 0; i < 3; i++) {
      int idx = t + i * 256;            // 768 float4
      int k = idx / 24, nq = idx % 24;
      float4 xv = *(const float4*)(xb + (size_t)(kc + k) * 9216 + p0 + nq * 4);
      int n = nq * 4;
      sB[(n + 0) * 40 + k] = (__bf16)xv.x;
      sB[(n + 1) * 40 + k] = (__bf16)xv.y;
      sB[(n + 2) * 40 + k] = (__bf16)xv.z;
      sB[(n + 3) * 40 + k] = (__bf16)xv.w;
    }
    if (kc + 32 < 512) {  // prefetch next x tile slice (global_prefetch_b8)
      int k = (t >> 5) + ((t & 31) >> 4) * 16, n = (t & 15) * 6;
      __builtin_prefetch(xb + (size_t)(kc + 32 + k) * 9216 + p0 + n, 0, 1);
    }
    __syncthreads();
    bf16x16 a = load_frag_A(sA + wv * 16 * 40, 40, 0, lane);
    bf16x16 bf[6];
    for (int nt = 0; nt < 6; nt++)
      bf[nt] = load_frag_B(sB + nt * 16 * 40, 40, 0, lane);
    for (int nt = 0; nt < 6; nt++)
      acc[nt] = wmma_bf16(a, bf[nt], acc[nt]);
  }

  const int half = lane >> 4, nl = lane & 15;
  for (int nt = 0; nt < 6; nt++) {
    AccU u; u.v = acc[nt];
    int p = p0 + nt * 16 + nl;
    for (int vvi = 0; vvi < 8; vvi++) {
      int r = m0 + wv * 16 + vvi + 8 * half;   // D row: M = v + 8*(lane/16)
      qkv[((size_t)b * 640 + r) * 9216 + p] = (__bf16)(u.f[vvi] + bcat[r]);
    }
  }
}

// ---------------------------------------------------------------------------
// Kernel 2: energies.  slice < 768: column mode (fixed w): E_H[h][j] = Q^T K.
//            slice >= 768: row mode (fixed h): E_W[w][j].
// 96x96 output, K=64. 288 threads = 9 waves, each wave a 2x2 tile group.
// ---------------------------------------------------------------------------
__global__ __launch_bounds__(288) void energy_k(
    const __bf16* __restrict__ qkv, float* __restrict__ e) {
  int s = blockIdx.x;
  bool colmode = (s < 768);
  if (!colmode) s -= 768;
  const int b = s / 96, f = s % 96;    // f = w (col mode) or h (row mode)

  __shared__ __bf16 sQ[96 * 72];       // [i][c], stride 72
  __shared__ __bf16 sK[96 * 72];

  const int t = threadIdx.x, lane = t & 31, wv = t >> 5;

  const __bf16* qb = qkv + (size_t)b * 640 * 9216;           // rows 0..63
  const __bf16* kb = qkv + ((size_t)b * 640 + 64) * 9216;    // rows 64..127

  for (int idx = t; idx < 6144; idx += 288) {
    int i = idx % 96, c = idx / 96;
    size_t off = (size_t)c * 9216 +
                 (colmode ? (size_t)i * 96 + f : (size_t)f * 96 + i);
    sQ[i * 72 + c] = qb[off];
    sK[i * 72 + c] = kb[off];
  }
  __syncthreads();

  const int gm = wv / 3, gn = wv % 3;
  const v8f zero = {0.f, 0.f, 0.f, 0.f, 0.f, 0.f, 0.f, 0.f};
  v8f acc[2][2];
  acc[0][0] = zero; acc[0][1] = zero; acc[1][0] = zero; acc[1][1] = zero;

  for (int kc = 0; kc < 64; kc += 32) {
    bf16x16 a0 = load_frag_A(sQ + (2 * gm + 0) * 16 * 72, 72, kc, lane);
    bf16x16 a1 = load_frag_A(sQ + (2 * gm + 1) * 16 * 72, 72, kc, lane);
    bf16x16 b0 = load_frag_B(sK + (2 * gn + 0) * 16 * 72, 72, kc, lane);
    bf16x16 b1 = load_frag_B(sK + (2 * gn + 1) * 16 * 72, 72, kc, lane);
    acc[0][0] = wmma_bf16(a0, b0, acc[0][0]);
    acc[0][1] = wmma_bf16(a0, b1, acc[0][1]);
    acc[1][0] = wmma_bf16(a1, b0, acc[1][0]);
    acc[1][1] = wmma_bf16(a1, b1, acc[1][1]);
  }

  const int half = lane >> 4, nl = lane & 15;
  for (int ai = 0; ai < 2; ai++)
    for (int bi = 0; bi < 2; bi++) {
      AccU u; u.v = acc[ai][bi];
      int n = (2 * gn + bi) * 16 + nl;
      for (int vvi = 0; vvi < 8; vvi++) {
        int m = (2 * gm + ai) * 16 + vvi + 8 * half;
        size_t o = colmode
            ? (((size_t)b * 96 + m) * 96 + f) * 192 + n          // E_H[b,h=m,w=f,j=n]
            : (((size_t)b * 96 + f) * 96 + m) * 192 + 96 + n;    // E_W[b,h=f,w=m,j=n]
        e[o] = u.f[vvi];
      }
    }
}

// ---------------------------------------------------------------------------
// Kernel 3: softmax over 192 energies per (b,h,w); one wave32 per row.
// Reads fp32, writes bf16 attention IN PLACE over the fp32 buffer (bf16 rows packed at stride 192).
// ---------------------------------------------------------------------------
__global__ __launch_bounds__(256) void softmax_k(float* __restrict__ e) {
  const int row  = blockIdx.x * 8 + (threadIdx.x >> 5);
  const int lane = threadIdx.x & 31;
  const size_t base = (size_t)row * 192;

  float v[6];
  float mx = -INFINITY;
  for (int i = 0; i < 6; i++) {
    v[i] = e[base + lane + 32 * i];
    mx = fmaxf(mx, v[i]);
  }
  for (int off = 16; off > 0; off >>= 1) mx = fmaxf(mx, __shfl_xor(mx, off, 32));
  float sum = 0.f;
  for (int i = 0; i < 6; i++) { v[i] = __expf(v[i] - mx); sum += v[i]; }
  for (int off = 16; off > 0; off >>= 1) sum += __shfl_xor(sum, off, 32);
  const float inv = 1.f / sum;

  __bf16* att = (__bf16*)e;            // bf16 row r at f32 offset 96*r (inside f32 row r/2)
  for (int i = 0; i < 6; i++) att[base + lane + 32 * i] = (__bf16)(v[i] * inv);
}

// ---------------------------------------------------------------------------
// Kernel 4: aggregation GEMMs, M=512 (4 blocks of 128), N=96, K=96.
// mode 0 (row):  out_W raw:      A[c][j]=v[b,c,h,j],  B[j][w]=att_W  (coalesced)
// mode 1 (col):  out = g*(out + out_H): A[c][j]=v[b,c,j,w], B[j][h]=att_H
// ---------------------------------------------------------------------------
__global__ __launch_bounds__(256) void out_gemm_k(
    const __bf16* __restrict__ qkv, const float* __restrict__ e,
    float* __restrict__ out, const float* __restrict__ gamma_p, int mode) {
  const __bf16* att = (const __bf16*)e;
  const int b = blockIdx.z, m0 = blockIdx.y * 128, f = blockIdx.x; // f = h|w
  const __bf16* vb = qkv + ((size_t)b * 640 + 128) * 9216;        // v rows

  __shared__ __bf16 sA[128 * 40];
  __shared__ __bf16 sB[96 * 40];

  const int t = threadIdx.x, lane = t & 31, wv = t >> 5;
  const v8f zero = {0.f, 0.f, 0.f, 0.f, 0.f, 0.f, 0.f, 0.f};
  v8f acc[6];
  for (int i = 0; i < 6; i++) acc[i] = zero;

  for (int kc = 0; kc < 96; kc += 32) {
    __syncthreads();
    if (mode == 0) {
      // A[m][k] = v[b, m0+m, h=f, j=kc+k]: contiguous k -> uint4 loads
      for (int i = 0; i < 2; i++) {
        int idx = t + i * 256;          // 512 uint4
        int m = idx >> 2, q4 = idx & 3;
        *(uint4*)(sA + m * 40 + q4 * 8) =
            *(const uint4*)(vb + (size_t)(m0 + m) * 9216 + (size_t)f * 96 + kc + q4 * 8);
      }
    } else {
      // A[m][k] = v[b, m0+m, j=kc+k, w=f]: strided gather (L2-resident)
      for (int i = 0; i < 16; i++) {
        int idx = t + i * 256;
        int m = idx >> 5, k = idx & 31;
        sA[m * 40 + k] = vb[(size_t)(m0 + m) * 9216 + (size_t)(kc + k) * 96 + f];
      }
    }
    // B tile: att rows contiguous over j -> uint4 loads (384 uint4)
    for (int idx = t; idx < 384; idx += 256) {
      int n = idx >> 2, q4 = idx & 3;
      size_t aoff = (mode == 0)
          ? (((size_t)b * 96 + f) * 96 + n) * 192 + 96 + kc + q4 * 8   // att_W
          : (((size_t)b * 96 + n) * 96 + f) * 192 + kc + q4 * 8;       // att_H
      *(uint4*)(sB + n * 40 + q4 * 8) = *(const uint4*)(att + aoff);
    }
    __syncthreads();
    bf16x16 a = load_frag_A(sA + wv * 16 * 40, 40, 0, lane);
    bf16x16 bf[6];
    for (int nt = 0; nt < 6; nt++)
      bf[nt] = load_frag_B(sB + nt * 16 * 40, 40, 0, lane);
    for (int nt = 0; nt < 6; nt++)
      acc[nt] = wmma_bf16(a, bf[nt], acc[nt]);
  }

  const float g = (mode == 0) ? 1.f : gamma_p[0];
  const int half = lane >> 4, nl = lane & 15;
  for (int nt = 0; nt < 6; nt++) {
    AccU u; u.v = acc[nt];
    int n = nt * 16 + nl;
    for (int vvi = 0; vvi < 8; vvi++) {
      int r = m0 + wv * 16 + vvi + 8 * half;
      size_t o = (mode == 0)
          ? ((size_t)b * 512 + r) * 9216 + (size_t)f * 96 + n    // [b,r,h=f,w=n]
          : ((size_t)b * 512 + r) * 9216 + (size_t)n * 96 + f;   // [b,r,h=n,w=f]
      if (mode == 0) out[o] = u.f[vvi];
      else           out[o] = g * (out[o] + u.f[vvi]);
    }
  }
}

// ---------------------------------------------------------------------------
// Host launcher
// ---------------------------------------------------------------------------
extern "C" void kernel_launch(void* const* d_in, const int* in_sizes, int n_in,
                              void* d_out, int out_size, void* d_ws, size_t ws_size,
                              hipStream_t stream) {
  const float* x     = (const float*)d_in[0];
  const float* Wq    = (const float*)d_in[1];
  const float* bq    = (const float*)d_in[2];
  const float* Wk    = (const float*)d_in[3];
  const float* bk    = (const float*)d_in[4];
  const float* Wv    = (const float*)d_in[5];
  const float* bv    = (const float*)d_in[6];
  const float* gamma = (const float*)d_in[7];
  float* out = (float*)d_out;

  // Workspace layout (bytes, 256-aligned):
  //   Wcat bf16 640*512            =    655,360  @ 0
  //   bcat f32  640                =      2,560  @ 655,360
  //   qkv  bf16 8*640*9216         = 94,371,840  @ 657,920
  //   e    f32  8*96*96*192        = 56,623,104  @ 95,029,760
  char* ws = (char*)d_ws;
  __bf16* Wcat = (__bf16*)(ws);
  float*  bcat = (float*) (ws + 655360);
  __bf16* qkv  = (__bf16*)(ws + 657920);
  float*  e    = (float*) (ws + 95029760);

  pack_weights_k<<<1280, 256, 0, stream>>>(Wq, Wk, Wv, bq, bk, bv, Wcat, bcat);
  proj_gemm_k<<<dim3(96, 5, 8), 256, 0, stream>>>(x, Wcat, bcat, qkv);
  energy_k<<<1536, 288, 0, stream>>>(qkv, e);
  softmax_k<<<9216, 256, 0, stream>>>(e);
  out_gemm_k<<<dim3(96, 4, 8), 256, 0, stream>>>(qkv, e, out, gamma, 0); // out_W
  out_gemm_k<<<dim3(96, 4, 8), 256, 0, stream>>>(qkv, e, out, gamma, 1); // + out_H, *gamma
}
